// GNNBranch1_34857954574435
// MI455X (gfx1250) — matrix operand order repaired and checked by reading, not compiled, not run on hardware
//
#include <hip/hip_runtime.h>
#include <hip/hip_bf16.h>
#include <math.h>

#define N_NODES   50000
#define N_EDGES   800000
#define F_IN      128
#define HEADS     4
#define HEAD_DIM  32
#define HIDDEN    128
#define N_GRAPHS  64
#define NMAT      512        // q|k|v|skip concatenated along N
#define LEAKY     0.01f
#define EPS_LN    1e-5f
#define INV_SQRT_HD 0.17677669529663687f   // 1/sqrt(32)

typedef __attribute__((ext_vector_type(16))) _Float16 v16h;
typedef __attribute__((ext_vector_type(8)))  _Float16 v8h;
typedef __attribute__((ext_vector_type(8)))  float    v8f;

// ---------------------------------------------------------------- utilities
__device__ __forceinline__ void atomicMaxF(float* addr, float v) {
    // works given init = -inf (0xFF800000): signed max handles v>=0,
    // unsigned min handles v<0 (more-negative floats have larger uint bits)
    if (v >= 0.0f) atomicMax((int*)addr, __float_as_int(v));
    else           atomicMin((unsigned int*)addr, (unsigned int)__float_as_int(v));
}

// ------------------------------------------------- 0. convert x -> f16
__global__ void cvt_x_f16(const float* __restrict__ x, _Float16* __restrict__ xh, int n) {
    int i = blockIdx.x * blockDim.x + threadIdx.x;
    if (i < n) xh[i] = (_Float16)x[i];
}

// ------------------------------------------------- 1. pack W (N-major, f16) + biases
__global__ void pack_weights(const float* __restrict__ wq, const float* __restrict__ wk,
                             const float* __restrict__ wv, const float* __restrict__ ws,
                             const float* __restrict__ bq, const float* __restrict__ bk,
                             const float* __restrict__ bv, const float* __restrict__ bs,
                             _Float16* __restrict__ wt, float* __restrict__ ball) {
    int i = blockIdx.x * blockDim.x + threadIdx.x;        // i = (mat*128 + n)*128 + k
    if (i >= 4 * F_IN * HIDDEN) return;
    int mat = i >> 14;
    int n   = (i >> 7) & 127;
    int k   = i & 127;
    const float* W = (mat == 0) ? wq : (mat == 1) ? wk : (mat == 2) ? wv : ws;
    wt[i] = (_Float16)W[k * HIDDEN + n];                  // transpose: K becomes contiguous
    if (k == 0) {
        const float* B = (mat == 0) ? bq : (mat == 1) ? bk : (mat == 2) ? bv : bs;
        ball[mat * HIDDEN + n] = B[n];
    }
}

// ------------------------------------------------- 2. init scratch each call
__global__ void init_ws(float* __restrict__ accum,   // [N_NODES*128] -> 0
                        float* __restrict__ mbuf,    // [N_NODES*4]   -> -inf
                        float* __restrict__ denom,   // [N_NODES*4]   -> 0
                        float* __restrict__ gstats)  // [192]         -> 0 (gsum|gvar|cnt)
{
    int i = blockIdx.x * blockDim.x + threadIdx.x;
    if (i < N_NODES * HIDDEN) accum[i] = 0.0f;
    if (i < N_NODES * HEADS) { mbuf[i] = -INFINITY; denom[i] = 0.0f; }
    if (i < 3 * N_GRAPHS) gstats[i] = 0.0f;
}

// ------------------------------------------------- 3. fused QKVS GEMM via WMMA
//  qkvs[50000][512] = xh[50000][128] @ Wall[128][512] + ball
//  one wave computes a 16x64 tile (4 accumulators, A-frag reused 4x);
//  8 waves/block cover all N=512; grid = 3125 M-tiles
__global__ __launch_bounds__(256)
void gemm_qkvs_wmma(const _Float16* __restrict__ xh,    // [N_NODES][128] row-major
                    const _Float16* __restrict__ wt,    // [512][128]  N-major (K contig)
                    const float*    __restrict__ ball,  // [512]
                    float*          __restrict__ qkvs)  // [N_NODES][512]
{
    const int lane  = threadIdx.x & 31;
    const int wave  = threadIdx.x >> 5;                 // 0..7  -> N cols [wave*64, wave*64+64)
    const int mtile = blockIdx.x;                       // 0..3124   (50000/16 = 3125)
    const int l16   = lane & 15;
    const int half  = lane >> 4;

    const int row   = mtile * 16 + l16;                 // A row  (M)
    const int ncol0 = wave * 64 + l16;                  // first B col (N) for j=0

    const _Float16* arow = xh + (size_t)row   * F_IN;
    const _Float16* brow = wt + (size_t)ncol0 * F_IN;   // j-th tile at +j*16*128 halves

    v8f acc[4] = {};
    union Frag { v16h v; v8h h[2]; };
#pragma unroll
    for (int kk = 0; kk < F_IN; kk += 32) {
        Frag a;
        // A 16x32 f16 layout: elems 0-7 = K[kk + 8*half .. +7], elems 8-15 = K[kk+16+8*half ..]
        a.h[0] = *(const v8h*)(arow + kk + 8 * half);
        a.h[1] = *(const v8h*)(arow + kk + 16 + 8 * half);
#pragma unroll
        for (int j = 0; j < 4; ++j) {
            Frag b;
            // B 32x16 f16 layout: elems 0-7 = K[kk + 8*half ..], elems 8-15 = K[kk+16+8*half ..] for column (ncol0 + 16j)
            const _Float16* bp = brow + (size_t)j * 16 * F_IN + kk + 8 * half;
            b.h[0] = *(const v8h*)(bp);
            b.h[1] = *(const v8h*)(bp + 16);
            acc[j] = __builtin_amdgcn_wmma_f32_16x16x32_f16(false, a.v, false, b.v,
                                                            (short)0, acc[j], false, false);
        }
    }
#pragma unroll
    for (int j = 0; j < 4; ++j) {
        const int ncol = ncol0 + j * 16;
        const float bias = ball[ncol];
#pragma unroll
        for (int r = 0; r < 8; ++r) {
            int m = mtile * 16 + r + 8 * half;          // D: VGPR r -> M = r + 8*half
            qkvs[(size_t)m * NMAT + ncol] = acc[j][r] + bias;
        }
    }
}

// ------------------------------------------------- 4. edge logits + segment max
__global__ void edge_logits(const int* __restrict__ ei, const float* __restrict__ qkvs,
                            float* __restrict__ alpha, float* __restrict__ mbuf) {
    int t = blockIdx.x * blockDim.x + threadIdx.x;
    if (t >= N_EDGES * HEADS) return;
    int e = t >> 2, h = t & 3;
    int src = ei[e];
    int dst = ei[N_EDGES + e];
    const float4* qp = (const float4*)(qkvs + (size_t)dst * NMAT + 0   + h * HEAD_DIM);
    const float4* kp = (const float4*)(qkvs + (size_t)src * NMAT + 128 + h * HEAD_DIM);
    float s = 0.0f;
#pragma unroll
    for (int i = 0; i < HEAD_DIM / 4; ++i) {
        float4 a = qp[i], b = kp[i];
        s += a.x * b.x + a.y * b.y + a.z * b.z + a.w * b.w;
    }
    s *= INV_SQRT_HD;
    alpha[(size_t)e * HEADS + h] = s;
    atomicMaxF(mbuf + (size_t)dst * HEADS + h, s);
}

// ------------------------------------------------- 5. exp + segment sum
//  (reference's where(isfinite(m), m, 0) folded in; -inf only occurs for
//   empty segments which no edge indexes, but keep exact semantics)
__global__ void edge_exp(const int* __restrict__ ei, const float* __restrict__ mbuf,
                         float* __restrict__ alpha, float* __restrict__ denom) {
    int t = blockIdx.x * blockDim.x + threadIdx.x;
    if (t >= N_EDGES * HEADS) return;
    int e = t >> 2, h = t & 3;
    int dst = ei[N_EDGES + e];
    float m = mbuf[(size_t)dst * HEADS + h];
    if (!isfinite(m)) m = 0.0f;
    float ex = __expf(alpha[(size_t)e * HEADS + h] - m);
    alpha[(size_t)e * HEADS + h] = ex;
    atomicAdd(denom + (size_t)dst * HEADS + h, ex);
}

// ------------------------------------------------- 6. attn + weighted scatter-add of v
__global__ void edge_scatter(const int* __restrict__ ei, const float* __restrict__ qkvs,
                             const float* __restrict__ alpha, const float* __restrict__ denom,
                             float* __restrict__ attn_out, float* __restrict__ accum) {
    int t = blockIdx.x * blockDim.x + threadIdx.x;
    if (t >= N_EDGES * HEADS) return;
    int e = t >> 2, h = t & 3;
    int src = ei[e];
    int dst = ei[N_EDGES + e];
    float a = alpha[(size_t)e * HEADS + h] / (denom[(size_t)dst * HEADS + h] + 1e-16f);
    attn_out[(size_t)e * HEADS + h] = a;
    const float* vp = qkvs + (size_t)src * NMAT + 256 + h * HEAD_DIM;
    float*       op = accum + (size_t)dst * HIDDEN + h * HEAD_DIM;
#pragma unroll
    for (int c = 0; c < HEAD_DIM; ++c) atomicAdd(op + c, a * vp[c]);
}

// ------------------------------------------------- 7. skip add + per-graph sum/count
//  one wave per node; lane c owns channels [4c,4c+4) as a float4 (coalesced)
__global__ void node_skip_stats(const float* __restrict__ qkvs, float* __restrict__ accum,
                                const int* __restrict__ batch,
                                float* __restrict__ gsum, float* __restrict__ cnt) {
    int t = blockIdx.x * blockDim.x + threadIdx.x;
    int n = t >> 5, lane = t & 31;
    if (n >= N_NODES) return;
    const float4* sp = (const float4*)(qkvs + (size_t)n * NMAT + 384);
    float4*       ap = (float4*)(accum + (size_t)n * HIDDEN);
    float4 v = ap[lane], s4 = sp[lane];
    v.x += s4.x; v.y += s4.y; v.z += s4.z; v.w += s4.w;
    ap[lane] = v;
    float s = v.x + v.y + v.z + v.w;
#pragma unroll
    for (int off = 16; off > 0; off >>= 1) s += __shfl_xor(s, off, 32);
    if (lane == 0) {
        int g = batch[n];
        atomicAdd(gsum + g, s);
        atomicAdd(cnt + g, 1.0f);
    }
}

// ------------------------------------------------- 8. per-graph variance (wave per node)
__global__ void node_var(const float* __restrict__ accum, const int* __restrict__ batch,
                         const float* __restrict__ gsum, const float* __restrict__ cnt,
                         float* __restrict__ gvar) {
    int t = blockIdx.x * blockDim.x + threadIdx.x;
    int n = t >> 5, lane = t & 31;
    if (n >= N_NODES) return;
    int g = batch[n];
    float norm = fmaxf(cnt[g], 1.0f) * (float)HIDDEN;
    float mean = gsum[g] / norm;
    const float4* ap = (const float4*)(accum + (size_t)n * HIDDEN);
    float4 v = ap[lane];
    float dx = v.x - mean, dy = v.y - mean, dz = v.z - mean, dw = v.w - mean;
    float s = dx * dx + dy * dy + dz * dz + dw * dw;
#pragma unroll
    for (int off = 16; off > 0; off >>= 1) s += __shfl_xor(s, off, 32);
    if (lane == 0) atomicAdd(gvar + g, s);
}

// ------------------------------------------------- 9. normalize + affine + LeakyReLU
__global__ void node_final(const float* __restrict__ accum, const int* __restrict__ batch,
                           const float* __restrict__ gsum, const float* __restrict__ cnt,
                           const float* __restrict__ gvar,
                           const float* __restrict__ lnw, const float* __restrict__ lnb,
                           float* __restrict__ y) {
    int t = blockIdx.x * blockDim.x + threadIdx.x;
    if (t >= N_NODES * HIDDEN) return;
    int n = t >> 7, c = t & 127;
    int g = batch[n];
    float norm = fmaxf(cnt[g], 1.0f) * (float)HIDDEN;
    float mean = gsum[g] / norm;
    float var  = gvar[g] / norm;
    float v = (accum[t] - mean) * rsqrtf(var + EPS_LN);
    v = v * lnw[c] + lnb[c];
    y[t] = (v > 0.0f) ? v : LEAKY * v;
}

// ================================================================ launch
extern "C" void kernel_launch(void* const* d_in, const int* in_sizes, int n_in,
                              void* d_out, int out_size, void* d_ws, size_t ws_size,
                              hipStream_t stream) {
    const float* x   = (const float*)d_in[0];
    const int*   ei  = (const int*)  d_in[1];
    const int*   bat = (const int*)  d_in[2];
    const float* Wq  = (const float*)d_in[3];
    const float* bq  = (const float*)d_in[4];
    const float* Wk  = (const float*)d_in[5];
    const float* bk  = (const float*)d_in[6];
    const float* Wv  = (const float*)d_in[7];
    const float* bv  = (const float*)d_in[8];
    const float* Ws  = (const float*)d_in[9];
    const float* bs  = (const float*)d_in[10];
    const float* lnw = (const float*)d_in[11];
    const float* lnb = (const float*)d_in[12];

    // output: y [50000*128] then attn [800000*4]
    float* y_out    = (float*)d_out;
    float* attn_out = y_out + (size_t)N_NODES * HIDDEN;

    // workspace carve-up (all chunks 256B-aligned sizes)
    char* p = (char*)d_ws;
    float*    qkvs  = (float*)p;            p += (size_t)N_NODES * NMAT * 4;    // 102.4 MB
    _Float16* xh    = (_Float16*)p;         p += (size_t)N_NODES * F_IN * 2;    // 12.8 MB
    _Float16* wt    = (_Float16*)p;         p += (size_t)NMAT * F_IN * 2;       // 128 KB
    float*    ball  = (float*)p;            p += (size_t)NMAT * 4;              // 2 KB
    float*    alpha = (float*)p;            p += (size_t)N_EDGES * HEADS * 4;   // 12.8 MB
    float*    mbuf  = (float*)p;            p += (size_t)N_NODES * HEADS * 4;   // 0.8 MB
    float*    denom = (float*)p;            p += (size_t)N_NODES * HEADS * 4;   // 0.8 MB
    float*    accum = (float*)p;            p += (size_t)N_NODES * HIDDEN * 4;  // 25.6 MB
    float*    gstats= (float*)p;            p += 256;                           // gsum|gvar|cnt
    float* gsum = gstats, *gvar = gstats + N_GRAPHS, *cnt = gstats + 2 * N_GRAPHS;

    const int B = 256;
    // 0/1/2: conversions + init
    cvt_x_f16 <<<(N_NODES * F_IN + B - 1) / B, B, 0, stream>>>(x, xh, N_NODES * F_IN);
    pack_weights<<<(4 * F_IN * HIDDEN + B - 1) / B, B, 0, stream>>>(Wq, Wk, Wv, Ws,
                                                                    bq, bk, bv, bs, wt, ball);
    init_ws   <<<(N_NODES * HIDDEN + B - 1) / B, B, 0, stream>>>(accum, mbuf, denom, gstats);

    // 3: fused q|k|v|skip GEMM via WMMA (3125 blocks; wave = 16x64 tile, 16 WMMAs)
    gemm_qkvs_wmma<<<N_NODES / 16, B, 0, stream>>>(xh, wt, ball, qkvs);

    // 4-6: edge pipeline
    const int ET = N_EDGES * HEADS;
    edge_logits <<<(ET + B - 1) / B, B, 0, stream>>>(ei, qkvs, alpha, mbuf);
    edge_exp    <<<(ET + B - 1) / B, B, 0, stream>>>(ei, mbuf, alpha, denom);
    edge_scatter<<<(ET + B - 1) / B, B, 0, stream>>>(ei, qkvs, alpha, denom, attn_out, accum);

    // 7-9: node pipeline (skip + graph LayerNorm + LeakyReLU), wave-per-node coalesced
    const int NT = N_NODES * 32;
    node_skip_stats<<<(NT + B - 1) / B, B, 0, stream>>>(qkvs, accum, bat, gsum, cnt);
    node_var       <<<(NT + B - 1) / B, B, 0, stream>>>(accum, bat, gsum, cnt, gvar);
    node_final     <<<(N_NODES * HIDDEN + B - 1) / B, B, 0, stream>>>(accum, bat, gsum, cnt,
                                                                      gvar, lnw, lnb, y_out);
}
